// GAT_L3_45801531245070
// MI455X (gfx1250) — matrix operand order repaired and checked by reading, not compiled, not run on hardware
//
#include <hip/hip_runtime.h>
#include <math.h>

// ---------------- problem constants (match reference) ----------------
#define N_NODES 50000
#define N_EDGES 800000
#define ETOT    (N_EDGES + N_NODES)   // self-loops appended
#define IN_DIM  128
#define HIDD    32
#define NHEADS  2
#define ODIM    32
#define BN_EPS  1e-5f
#define LSLOPE  0.2f

typedef __bf16 bf16;
typedef bf16  v16bf __attribute__((ext_vector_type(16)));
typedef bf16  v8bf  __attribute__((ext_vector_type(8)));
typedef float v8f   __attribute__((ext_vector_type(8)));

// ---------------- f32 -> bf16 conversion (activations) ----------------
__global__ void cvt_f32_bf16(const float* __restrict__ X, bf16* __restrict__ Y, int n) {
  for (int i = blockIdx.x * blockDim.x + threadIdx.x; i < n;
       i += gridDim.x * blockDim.x)
    Y[i] = (bf16)X[i];
}

// ---------------- f32 W[K,NC] -> bf16 Wt[NC,K] (transpose) -----------
__global__ void cvt_w_bf16_t(const float* __restrict__ W, bf16* __restrict__ Wt,
                             int K, int NC) {
  int total = K * NC;
  for (int i = blockIdx.x * blockDim.x + threadIdx.x; i < total;
       i += gridDim.x * blockDim.x) {
    int k = i / NC, c = i - k * NC;
    Wt[(size_t)c * K + k] = (bf16)W[i];
  }
}

// ---------------------------------------------------------------------
// H[N,NC] = A[N,K] * W[K,NC], A in bf16 row-major, W pre-transposed to
// bf16 Wt[NC,K]. One wave owns a full 16-row x NC stripe: one shared A
// fragment per k-step feeds NC/16 independent WMMA accumulator chains.
// Fully unrolled (K, NC compile-time) -> pipelined v_wmma stream with
// 128-bit fragment loads only.
//
// Fragment layouts (wave32):
//  A 16x32 bf16 : lane<16 rows 0..15 K[0..7]+K[16..23]; lane>=16 K[8..15]+K[24..31]
//  B 32x16 bf16 : col=lane%16; lane<16 K[0..7]+K[16..23]; lane>=16 K[8..15]+K[24..31]
//  C/D 16x16 f32: vgpr r -> M=r (lanes 0-15) / M=8+r (lanes 16-31), N=lane%16
// ---------------------------------------------------------------------
template <int K, int NC>
__global__ void gemm_wmma(const bf16* __restrict__ A,
                          const bf16* __restrict__ Wt,
                          float* __restrict__ H) {
  constexpr int TN = NC / 16;
  const int lane = threadIdx.x & 31;
  const int wave = blockIdx.x * (blockDim.x >> 5) + (threadIdx.x >> 5);
  if (wave >= N_NODES / 16) return;           // wave-uniform: EXEC stays full
  const int half = lane >> 4;                 // 0 or 1
  const int l16  = lane & 15;
  const bf16* Ap = A + (size_t)(wave * 16 + l16) * K + half * 8;

  v8f acc[TN];
#pragma unroll
  for (int t = 0; t < TN; ++t) acc[t] = (v8f){};

#pragma unroll
  for (int kb = 0; kb < K; kb += 32) {
    union { v16bf v; v8bf h[2]; } a;
    a.h[0] = *(const v8bf*)(Ap + kb);         // K = kb+half*8 .. +7
    a.h[1] = *(const v8bf*)(Ap + kb + 16);    // K = kb+16+half*8 .. +7
#pragma unroll
    for (int t = 0; t < TN; ++t) {
      const bf16* Wp = Wt + (size_t)(t * 16 + l16) * K + kb + half * 8;
      union { v16bf v; v8bf h[2]; } b;
      b.h[0] = *(const v8bf*)(Wp);            // K = kb+half*8 .. +7
      b.h[1] = *(const v8bf*)(Wp + 16);       // K = kb+16+half*8 .. +7
      acc[t] = __builtin_amdgcn_wmma_f32_16x16x32_bf16(false, a.v, false, b.v,
                                                       (short)0, acc[t],
                                                       false, false);
    }
  }
#pragma unroll
  for (int t = 0; t < TN; ++t) {
    float* Hp = H + (size_t)(wave * 16) * NC + t * 16 + l16;
#pragma unroll
    for (int r = 0; r < 8; ++r)
      Hp[(size_t)(r + half * 8) * NC] = acc[t][r];
  }
}

// ---------------- per-node attention coefficients --------------------
__global__ void attn_coeff(const float* __restrict__ h,
                           const float* __restrict__ a_src,
                           const float* __restrict__ a_dst,
                           float* __restrict__ esrc, float* __restrict__ edst,
                           int Hh, int D) {
  int total = N_NODES * Hh;
  for (int i = blockIdx.x * blockDim.x + threadIdx.x; i < total;
       i += gridDim.x * blockDim.x) {
    int nid = i / Hh, hd = i - nid * Hh;
    const float* hp = h + (size_t)nid * Hh * D + hd * D;
    const float* as = a_src + hd * D;
    const float* ad = a_dst + hd * D;
    float s = 0.f, t = 0.f;
#pragma unroll 8
    for (int d = 0; d < D; ++d) { s += hp[d] * as[d]; t += hp[d] * ad[d]; }
    esrc[i] = s; edst[i] = t;
  }
}

// ---------------- generic fill ---------------------------------------
__global__ void fill_f32(float* __restrict__ p, float v, int n) {
  for (int i = blockIdx.x * blockDim.x + threadIdx.x; i < n;
       i += gridDim.x * blockDim.x)
    p[i] = v;
}

// ---------------- float atomic max via int/uint trick -----------------
__device__ __forceinline__ void atomicMaxF(float* addr, float value) {
  if (value >= 0.f) atomicMax((int*)addr, __float_as_int(value));
  else              atomicMin((unsigned int*)addr, __float_as_uint(value));
}

// ---------------- edge pass A: leaky-relu logits + segment max --------
__global__ void edge_logits_max(const int* __restrict__ ei,
                                const float* __restrict__ esrc,
                                const float* __restrict__ edst,
                                float* __restrict__ ebuf,
                                float* __restrict__ emax, int Hh) {
  int total = ETOT * Hh;
  for (int i = blockIdx.x * blockDim.x + threadIdx.x; i < total;
       i += gridDim.x * blockDim.x) {
    int eid = i / Hh, hd = i - eid * Hh;
    int s = (eid < N_EDGES) ? ei[eid]           : (eid - N_EDGES);
    int d = (eid < N_EDGES) ? ei[N_EDGES + eid] : (eid - N_EDGES);
    float v = esrc[s * Hh + hd] + edst[d * Hh + hd];
    v = (v > 0.f) ? v : LSLOPE * v;
    ebuf[i] = v;
    atomicMaxF(&emax[d * Hh + hd], v);
  }
}

// ---------------- edge pass B: exp + segment sum ----------------------
__global__ void edge_exp_sum(const int* __restrict__ ei,
                             float* __restrict__ ebuf,
                             const float* __restrict__ emax,
                             float* __restrict__ denom, int Hh) {
  int total = ETOT * Hh;
  for (int i = blockIdx.x * blockDim.x + threadIdx.x; i < total;
       i += gridDim.x * blockDim.x) {
    int eid = i / Hh, hd = i - eid * Hh;
    int d = (eid < N_EDGES) ? ei[N_EDGES + eid] : (eid - N_EDGES);
    float ex = __expf(ebuf[i] - emax[d * Hh + hd]);
    ebuf[i] = ex;
    atomicAdd(&denom[d * Hh + hd], ex);
  }
}

// ---------------- edge pass C: weighted scatter-aggregate -------------
// one thread per (edge, channel): gather h[src,c], atomic-add agg[dst,c]
__global__ void edge_aggregate(const int* __restrict__ ei,
                               const float* __restrict__ ebuf,
                               const float* __restrict__ denom,
                               const float* __restrict__ h,
                               float* __restrict__ agg, int Hh, int D) {
  int HD = Hh * D;
  long long total = (long long)ETOT * HD;
  for (long long i = (long long)blockIdx.x * blockDim.x + threadIdx.x;
       i < total; i += (long long)gridDim.x * blockDim.x) {
    int eid = (int)(i / HD);
    int c   = (int)(i - (long long)eid * HD);
    int hd  = c / D;
    int s = (eid < N_EDGES) ? ei[eid]           : (eid - N_EDGES);
    int d = (eid < N_EDGES) ? ei[N_EDGES + eid] : (eid - N_EDGES);
    float alpha = ebuf[eid * Hh + hd] / (denom[d * Hh + hd] + 1e-16f);
    atomicAdd(&agg[(size_t)d * HD + c], alpha * h[(size_t)s * HD + c]);
  }
}

// ---------------- BatchNorm stats: one block per channel --------------
// folds bias/gamma/beta into fused scale/shift. LDS tree reduction.
__global__ void bn_stats(const float* __restrict__ X,
                         const float* __restrict__ bias,
                         const float* __restrict__ gamma,
                         const float* __restrict__ beta,
                         float* __restrict__ scale,
                         float* __restrict__ shift, int n, int C) {
  const int c = blockIdx.x;
  __shared__ float ssum[256];
  __shared__ float ssq[256];
  const float bc = bias[c];
  float s = 0.f, q = 0.f;
  for (int i = threadIdx.x; i < n; i += blockDim.x) {
    float v = X[(size_t)i * C + c] + bc;
    s += v; q += v * v;
  }
  ssum[threadIdx.x] = s; ssq[threadIdx.x] = q;
  __syncthreads();
  for (int off = 128; off > 0; off >>= 1) {
    if (threadIdx.x < off) {
      ssum[threadIdx.x] += ssum[threadIdx.x + off];
      ssq[threadIdx.x]  += ssq[threadIdx.x + off];
    }
    __syncthreads();
  }
  if (threadIdx.x == 0) {
    float mu   = ssum[0] / (float)n;
    float var  = ssq[0] / (float)n - mu * mu;
    float rstd = rsqrtf(var + BN_EPS);
    float sc   = gamma[c] * rstd;
    scale[c] = sc;
    shift[c] = beta[c] - mu * sc;
  }
}

// ---------------- fused bias + BN + ReLU ------------------------------
__global__ void bn_apply_relu(const float* __restrict__ X,
                              const float* __restrict__ bias,
                              const float* __restrict__ scale,
                              const float* __restrict__ shift,
                              float* __restrict__ Y, int n, int C) {
  int total = n * C;
  for (int i = blockIdx.x * blockDim.x + threadIdx.x; i < total;
       i += gridDim.x * blockDim.x) {
    int c = i % C;
    float v = (X[i] + bias[c]) * scale[c] + shift[c];
    Y[i] = (v > 0.f) ? v : 0.f;
  }
}

// ---------------------------------------------------------------------
static void run_gat_layer(const float* X, int K,
                          const float* W, const float* as, const float* ad,
                          const float* bias, const float* gamma, const float* beta,
                          int Hh, int D, float* Y, const int* ei,
                          float* h, float* agg, float* esrc, float* edst,
                          float* emax, float* denom, float* ebuf,
                          float* scale, float* shift,
                          bf16* Abf, bf16* Wt, hipStream_t stream) {
  const int HD = Hh * D;
  // 1) dense projection via WMMA (bf16 operands, f32 accumulate)
  cvt_f32_bf16<<<2048, 256, 0, stream>>>(X, Abf, N_NODES * K);
  cvt_w_bf16_t<<<32, 256, 0, stream>>>(W, Wt, K, HD);
  {
    int blocks = (N_NODES / 16 + 7) / 8;          // 8 waves / 256-thread block
    if (K == IN_DIM && HD == 64)
      gemm_wmma<IN_DIM, 64><<<blocks, 256, 0, stream>>>(Abf, Wt, h);
    else if (K == 64 && HD == 64)
      gemm_wmma<64, 64><<<blocks, 256, 0, stream>>>(Abf, Wt, h);
    else
      gemm_wmma<64, 32><<<blocks, 256, 0, stream>>>(Abf, Wt, h);
  }
  // 2) per-node attention coefficients
  attn_coeff<<<512, 256, 0, stream>>>(h, as, ad, esrc, edst, Hh, D);
  // 3) init segment buffers
  fill_f32<<<256, 256, 0, stream>>>(emax, -INFINITY, N_NODES * Hh);
  fill_f32<<<256, 256, 0, stream>>>(denom, 0.f, N_NODES * Hh);
  fill_f32<<<2048, 256, 0, stream>>>(agg, 0.f, N_NODES * HD);
  // 4) segment softmax + aggregation
  edge_logits_max<<<2048, 256, 0, stream>>>(ei, esrc, edst, ebuf, emax, Hh);
  edge_exp_sum<<<2048, 256, 0, stream>>>(ei, ebuf, emax, denom, Hh);
  edge_aggregate<<<8192, 256, 0, stream>>>(ei, ebuf, denom, h, agg, Hh, D);
  // 5) BN (training mode) + ReLU, bias folded in
  bn_stats<<<HD, 256, 0, stream>>>(agg, bias, gamma, beta, scale, shift, N_NODES, HD);
  bn_apply_relu<<<2048, 256, 0, stream>>>(agg, bias, scale, shift, Y, N_NODES, HD);
}

extern "C" void kernel_launch(void* const* d_in, const int* in_sizes, int n_in,
                              void* d_out, int out_size, void* d_ws, size_t ws_size,
                              hipStream_t stream) {
  // inputs in setup_inputs() order
  const float* x   = (const float*)d_in[0];
  const float* W1  = (const float*)d_in[1];
  const float* as1 = (const float*)d_in[2];
  const float* ad1 = (const float*)d_in[3];
  const float* b1  = (const float*)d_in[4];
  const float* g1  = (const float*)d_in[5];
  const float* be1 = (const float*)d_in[6];
  const float* W2  = (const float*)d_in[7];
  const float* as2 = (const float*)d_in[8];
  const float* ad2 = (const float*)d_in[9];
  const float* b2  = (const float*)d_in[10];
  const float* g2  = (const float*)d_in[11];
  const float* be2 = (const float*)d_in[12];
  const float* W3  = (const float*)d_in[13];
  const float* as3 = (const float*)d_in[14];
  const float* ad3 = (const float*)d_in[15];
  const float* b3  = (const float*)d_in[16];
  const float* g3  = (const float*)d_in[17];
  const float* be3 = (const float*)d_in[18];
  const int*   ei  = (const int*)d_in[19];     // [2, E] row-major: src row, dst row
  float* out = (float*)d_out;                  // [N, 32]

  // workspace layout (~73 MB); float prefix is 16B-aligned in total size
  float* f     = (float*)d_ws;
  float* h     = f; f += (size_t)N_NODES * 64;
  float* agg   = f; f += (size_t)N_NODES * 64;
  float* buf1  = f; f += (size_t)N_NODES * 64;
  float* buf2  = f; f += (size_t)N_NODES * 64;
  float* esrc  = f; f += (size_t)N_NODES * NHEADS;
  float* edst  = f; f += (size_t)N_NODES * NHEADS;
  float* emax  = f; f += (size_t)N_NODES * NHEADS;
  float* denom = f; f += (size_t)N_NODES * NHEADS;
  float* ebuf  = f; f += (size_t)ETOT * NHEADS;
  float* scale = f; f += 64;
  float* shift = f; f += 64;
  bf16*  Abf   = (bf16*)f;                       // N * 128 bf16 (max)
  bf16*  Wt    = Abf + (size_t)N_NODES * IN_DIM; // 128 * 64 bf16 (max)

  // layer 1: [N,128] -> GAT(2 heads x 32, concat) -> BN -> ReLU -> buf1 [N,64]
  run_gat_layer(x, IN_DIM, W1, as1, ad1, b1, g1, be1, NHEADS, HIDD, buf1, ei,
                h, agg, esrc, edst, emax, denom, ebuf, scale, shift, Abf, Wt, stream);
  // layer 2: [N,64] -> GAT(2 heads x 32, concat) -> BN -> ReLU -> buf2 [N,64]
  run_gat_layer(buf1, NHEADS * HIDD, W2, as2, ad2, b2, g2, be2, NHEADS, HIDD, buf2, ei,
                h, agg, esrc, edst, emax, denom, ebuf, scale, shift, Abf, Wt, stream);
  // layer 3: [N,64] -> GAT(1 head x 32, mean==identity) -> BN -> ReLU -> out [N,32]
  run_gat_layer(buf2, NHEADS * HIDD, W3, as3, ad3, b3, g3, be3, 1, ODIM, out, ei,
                h, agg, esrc, edst, emax, denom, ebuf, scale, shift, Abf, Wt, stream);
}
